// att_deocder_4861902979557
// MI455X (gfx1250) — matrix operand order repaired and checked by reading, not compiled, not run on hardware
//
#include <hip/hip_runtime.h>
#include <hip/hip_bf16.h>
#include <cstdint>

#define BS    32
#define TS    2048
#define VDIMC 512
#define QDIM  1024
#define ADIM  256
#define NH    4
#define KD    512        // inner dim of k/v projections
#define NKCOL 1024       // NH*ADIM
#define NVCOL 2048       // NH*VDIMC

typedef __attribute__((ext_vector_type(16))) __bf16 v16bf;
typedef __attribute__((ext_vector_type(8)))  __bf16 v8bf;
typedef __attribute__((ext_vector_type(8)))  float  v8f;

__device__ __forceinline__ unsigned short f2bf(float f) {
    unsigned u = __builtin_bit_cast(unsigned, f);
    u += 0x7FFFu + ((u >> 16) & 1u);               // round-to-nearest-even
    return (unsigned short)(u >> 16);
}
__device__ __forceinline__ float bf2f(unsigned short s) {
    return __builtin_bit_cast(float, (unsigned)s << 16);
}
__device__ __forceinline__ float fast_tanh(float x) {
    float e = __expf(2.0f * x);
    return 1.0f - 2.0f * __builtin_amdgcn_rcpf(e + 1.0f);
}

// A-fragment for v_wmma_f32_16x16x32_bf16 from an LDS tile [16 rows][512 K] (bf16).
// Lane layout (contiguous, matching B): lanes 0-15 row=lane, K = {0..15};
// lanes 16-31 row=lane-16, K = {16..31}  => k0 = kb*32 + (lane>=16 ? 16 : 0).
__device__ __forceinline__ v16bf load_a(const unsigned short* lds, int row, int k0) {
    const v8bf lo = *reinterpret_cast<const v8bf*>(lds + row * KD + k0);
    const v8bf hi = *reinterpret_cast<const v8bf*>(lds + row * KD + k0 + 8);
    v16bf a;
#pragma unroll
    for (int i = 0; i < 8; i++) { a[i] = lo[i]; a[i + 8] = hi[i]; }
    return a;
}

// ---------------------------------------------------------------- prep: fp32 -> bf16 weights
__global__ void prep_kernel(const float* __restrict__ Wk, const float* __restrict__ Wv,
                            unsigned short* __restrict__ wkb, unsigned short* __restrict__ wvb) {
    const int n1 = NKCOL * KD;
    const int n2 = NVCOL * KD;
    for (int i = blockIdx.x * blockDim.x + threadIdx.x; i < n1 + n2; i += gridDim.x * blockDim.x) {
        if (i < n1) wkb[i] = f2bf(Wk[i]);
        else        wvb[i - n1] = f2bf(Wv[i - n1]);
    }
}

// ---------------------------------------------------------------- q = tanh(dec @ Wq^T + bq)
__global__ __launch_bounds__(256) void q_kernel(const float* __restrict__ dec,
                                                const float* __restrict__ Wq,
                                                const float* __restrict__ bq,
                                                float* __restrict__ q) {
    const int idx = blockIdx.x * 256 + threadIdx.x;     // 32768 outputs
    const int b = idx >> 10, col = idx & 1023;
    const float* x = dec + (size_t)b * QDIM;
    const float* w = Wq + (size_t)col * QDIM;
    float acc = 0.0f;
    for (int k = 0; k < QDIM; k += 4) {
        const float4 xv = *(const float4*)(x + k);
        const float4 wv = *(const float4*)(w + k);
        acc += xv.x * wv.x + xv.y * wv.y + xv.z * wv.z + xv.w * wv.w;
    }
    q[idx] = fast_tanh(acc + bq[col]);
}

// ---------------------------------------------------------------- loc_ctx (bf16)
// prev_att is uniform 1/len over valid frames => conv reduces to a windowed sum of
// Wsum[c][j] = sum_h Wconv[c][h][j].  loc_ctx[b,t,d] = tanh(sum_c Wloc[d,c]*loc[b,c,t]).
__global__ __launch_bounds__(256) void loc_kernel(const float* __restrict__ Wconv,  // [64][4][31]
                                                  const float* __restrict__ Wloc,   // [256][64]
                                                  const int* __restrict__ enc_len,
                                                  unsigned short* __restrict__ locb) {
    const int tid = threadIdx.x;
    const int b = blockIdx.x >> 7;
    const int tbase = (blockIdx.x & 127) << 4;
    const int len = enc_len[b];
    const float inv = 1.0f / (float)len;

    __shared__ float loc_s[16][64];

#pragma unroll
    for (int e = 0; e < 4; e++) {
        const int li = tid * 4 + e;
        const int r = li >> 6;
        const int c = li & 63;
        const int t = tbase + r;
        float s = 0.0f;
        for (int j = 0; j < 31; j++) {
            const int tp = t + j - 15;
            if (tp >= 0 && tp < len) {
                const float* w = Wconv + (size_t)c * 4 * 31 + j;
                s += w[0] + w[31] + w[62] + w[93];
            }
        }
        loc_s[r][c] = s * inv;
    }
    __syncthreads();

    const int r = tid >> 4;
    const int dg = (tid & 15) * 16;
    unsigned short out[16];
#pragma unroll
    for (int e = 0; e < 16; e++) {
        const float* wl = Wloc + (size_t)(dg + e) * 64;
        float acc = 0.0f;
#pragma unroll
        for (int c = 0; c < 64; c++) acc += wl[c] * loc_s[r][c];
        out[e] = f2bf(fast_tanh(acc));
    }
    unsigned short* dst = locb + ((size_t)b * TS + tbase + r) * ADIM + dg;
#pragma unroll
    for (int e = 0; e < 4; e++) {
        ushort4 o; o.x = out[e*4]; o.y = out[e*4+1]; o.z = out[e*4+2]; o.w = out[e*4+3];
        *(ushort4*)(dst + e * 4) = o;
    }
}

// ---------------------------------------------------------------- Pass A: fused k-proj + energy
// grid = 32 b * 128 t-tiles; block = 256 (8 waves). Each wave owns 8 N-tiles (128 k-cols).
__global__ __launch_bounds__(256) void energy_kernel(const float* __restrict__ enc,
                                                     const unsigned short* __restrict__ wkb,
                                                     const float* __restrict__ bk,
                                                     const float* __restrict__ q,
                                                     const unsigned short* __restrict__ locb,
                                                     const float* __restrict__ We,
                                                     const float* __restrict__ be,
                                                     float* __restrict__ energy) {
    const int tid = threadIdx.x;
    const int b = blockIdx.x >> 7;
    const int tbase = (blockIdx.x & 127) << 4;

    __shared__ __align__(16) unsigned short encb[16 * KD];
    __shared__ float eacc[NH * 16];

    if (tid < NH * 16) eacc[tid] = 0.0f;

    // fp32 -> bf16 tile into LDS (16 rows x 512)
    const float* src = enc + ((size_t)b * TS + tbase) * KD;
    {
        const int base = tid * 32;
#pragma unroll
        for (int c = 0; c < 8; c++) {
            const float4 f = *(const float4*)(src + base + c * 4);
            ushort4 o; o.x = f2bf(f.x); o.y = f2bf(f.y); o.z = f2bf(f.z); o.w = f2bf(f.w);
            *(ushort4*)(encb + base + c * 4) = o;
        }
    }
    __syncthreads();

    const int wave = tid >> 5;
    const int lane = tid & 31;
    const int l16 = lane & 15;
    const int hs = lane >> 4;

    const unsigned short* wbase = wkb + (size_t)(wave * 128 + l16) * KD + hs * 16;

    v8f acc[8] = {};
    for (int kb = 0; kb < 16; kb++) {
        const v16bf a = load_a(encb, l16, kb * 32 + hs * 16);
        // batch all 8 B-fragments first so the compiler can issue partial
        // s_wait_loadcnt and overlap later loads with earlier WMMAs
        v16bf bf[8];
#pragma unroll
        for (int n = 0; n < 8; n++) {
            // B lane layout: lanes 0-15 col=lane K=kb*32+0..15, lanes 16-31 col=lane-16 K=+16..31
            bf[n] = *reinterpret_cast<const v16bf*>(wbase + (size_t)n * 16 * KD + kb * 32);
        }
#pragma unroll
        for (int n = 0; n < 8; n++) {
            acc[n] = __builtin_amdgcn_wmma_f32_16x16x32_bf16(false, a, false, bf[n],
                                                             (short)0, acc[n], false, false);
        }
    }

#pragma unroll
    for (int n = 0; n < 8; n++) {
        const int cb = wave * 128 + n * 16;
        const int ncol = cb + l16;
        const int h = ncol >> 8;
        const int d = ncol & 255;
        const float wed = We[d];
        const float qq = q[(size_t)b * NKCOL + ncol];
        const float bkc = bk[ncol];
        const size_t lbase = ((size_t)b * TS + tbase + hs * 8) * ADIM + d;
#pragma unroll
        for (int r = 0; r < 8; r++) {
            const float kv = fast_tanh(acc[n][r] + bkc);                 // k = tanh(proj)
            const float lv = bf2f(locb[lbase + (size_t)r * ADIM]);
            float s = fast_tanh(kv + qq + lv) * wed;                     // e = tanh(k+q+loc)
            s += __shfl_xor(s, 1, 16);
            s += __shfl_xor(s, 2, 16);
            s += __shfl_xor(s, 4, 16);
            s += __shfl_xor(s, 8, 16);
            if (l16 == 0) atomicAdd(&eacc[h * 16 + hs * 8 + r], s);
        }
    }
    __syncthreads();
    if (tid < NH * 16) {
        const int h = tid >> 4, r = tid & 15;
        energy[((size_t)b * NH + h) * TS + tbase + r] = (eacc[tid] + be[0]) * (1.0f / 16.0f);
    }
}

// ---------------------------------------------------------------- masked softmax over t
__global__ __launch_bounds__(256) void softmax_kernel(const float* __restrict__ energy,
                                                      const int* __restrict__ enc_len,
                                                      float* __restrict__ attn) {
    const int bh = blockIdx.x;
    const int len = enc_len[bh >> 2];
    const int tid = threadIdx.x;
    const float* e = energy + (size_t)bh * TS;
    __shared__ float red[256];

    float v[8];
    float m = -3.0e38f;
#pragma unroll
    for (int i = 0; i < 8; i++) {
        const int t = tid + i * 256;
        const float x = (t < len) ? e[t] : -1.0e30f;
        v[i] = x; m = fmaxf(m, x);
    }
    red[tid] = m; __syncthreads();
    for (int s = 128; s > 0; s >>= 1) { if (tid < s) red[tid] = fmaxf(red[tid], red[tid + s]); __syncthreads(); }
    m = red[0]; __syncthreads();

    float sum = 0.0f;
#pragma unroll
    for (int i = 0; i < 8; i++) { v[i] = __expf(v[i] - m); sum += v[i]; }
    red[tid] = sum; __syncthreads();
    for (int s = 128; s > 0; s >>= 1) { if (tid < s) red[tid] += red[tid + s]; __syncthreads(); }
    const float inv = 1.0f / red[0];
#pragma unroll
    for (int i = 0; i < 8; i++) attn[(size_t)bh * TS + tid + i * 256] = v[i] * inv;
}

// ---------------------------------------------------------------- Pass B: fused v-proj + attn·v
// grid = 32 b * 8 column-splits (256 v-cols each); 8 waves * 2 N-tiles/wave; loop all t.
__global__ __launch_bounds__(256) void ctx_kernel(const float* __restrict__ enc,
                                                  const unsigned short* __restrict__ wvb,
                                                  const float* __restrict__ bv,
                                                  const float* __restrict__ attn,
                                                  float* __restrict__ ctx) {
    const int tid = threadIdx.x;
    const int b = blockIdx.x >> 3;
    const int cs = blockIdx.x & 7;
    const int h = cs >> 1;                         // 256 cols fit inside one head (512 wide)

    __shared__ __align__(16) unsigned short encb[16 * KD];
    __shared__ float attn_s[16];

    const int wave = tid >> 5;
    const int lane = tid & 31;
    const int l16 = lane & 15;
    const int hs = lane >> 4;

    const int cb0 = cs * 256 + wave * 32;
    const int cb1 = cb0 + 16;
    const float bv0 = bv[cb0 + l16];
    const float bv1 = bv[cb1 + l16];
    const float* arow = attn + ((size_t)b * NH + h) * TS;
    const unsigned short* wb0 = wvb + (size_t)(cb0 + l16) * KD + hs * 16;
    const unsigned short* wb1 = wvb + (size_t)(cb1 + l16) * KD + hs * 16;

    float c0 = 0.0f, c1 = 0.0f;

    for (int tc = 0; tc < 128; tc++) {
        const int tbase = tc * 16;
        __syncthreads();
        const float* src = enc + ((size_t)b * TS + tbase) * KD;
        {
            const int base = tid * 32;
#pragma unroll
            for (int c = 0; c < 8; c++) {
                const float4 f = *(const float4*)(src + base + c * 4);
                ushort4 o; o.x = f2bf(f.x); o.y = f2bf(f.y); o.z = f2bf(f.z); o.w = f2bf(f.w);
                *(ushort4*)(encb + base + c * 4) = o;
            }
        }
        if (tid < 16) attn_s[tid] = arow[tbase + tid];
        __syncthreads();

        v8f d0 = {}, d1 = {};
        for (int kb = 0; kb < 16; kb++) {
            const v16bf a = load_a(encb, l16, kb * 32 + hs * 16);
            const v16bf bf0 = *reinterpret_cast<const v16bf*>(wb0 + kb * 32);
            const v16bf bf1 = *reinterpret_cast<const v16bf*>(wb1 + kb * 32);
            d0 = __builtin_amdgcn_wmma_f32_16x16x32_bf16(false, a, false, bf0, (short)0, d0, false, false);
            d1 = __builtin_amdgcn_wmma_f32_16x16x32_bf16(false, a, false, bf1, (short)0, d1, false, false);
        }
#pragma unroll
        for (int r = 0; r < 8; r++) {
            const float aw = attn_s[hs * 8 + r];
            c0 += aw * fast_tanh(d0[r] + bv0);    // v = tanh(proj + bv)
            c1 += aw * fast_tanh(d1[r] + bv1);
        }
    }
    c0 += __shfl_xor(c0, 16, 32);                 // combine rows 0-7 / 8-15 halves
    c1 += __shfl_xor(c1, 16, 32);
    if (lane < 16) {
        ctx[(size_t)b * NVCOL + cb0 + lane] = c0;
        ctx[(size_t)b * NVCOL + cb1 + lane] = c1;
    }
}

// ---------------------------------------------------------------- context = ctx @ Wm^T + bm
__global__ __launch_bounds__(256) void merge_kernel(const float* __restrict__ ctx,
                                                    const float* __restrict__ Wm,
                                                    const float* __restrict__ bm,
                                                    float* __restrict__ out) {
    const int idx = blockIdx.x * 256 + threadIdx.x;  // 16384 outputs
    const int b = idx >> 9, o = idx & 511;
    const float* x = ctx + (size_t)b * NVCOL;
    const float* w = Wm + (size_t)o * NVCOL;
    float acc = 0.0f;
    for (int k = 0; k < NVCOL; k += 4) {
        const float4 xv = *(const float4*)(x + k);
        const float4 wv = *(const float4*)(w + k);
        acc += xv.x * wv.x + xv.y * wv.y + xv.z * wv.z + xv.w * wv.w;
    }
    out[idx] = acc + bm[o];
}

extern "C" void kernel_launch(void* const* d_in, const int* in_sizes, int n_in,
                              void* d_out, int out_size, void* d_ws, size_t ws_size,
                              hipStream_t stream) {
    (void)in_sizes; (void)n_in; (void)out_size; (void)ws_size;
    const float* dec   = (const float*)d_in[0];
    const float* enc   = (const float*)d_in[1];
    const float* Wq    = (const float*)d_in[2];
    const float* bq    = (const float*)d_in[3];
    const float* Wk    = (const float*)d_in[4];
    const float* bk    = (const float*)d_in[5];
    const float* Wv    = (const float*)d_in[6];
    const float* bv    = (const float*)d_in[7];
    const float* Wconv = (const float*)d_in[8];
    const float* Wloc  = (const float*)d_in[9];
    const float* We    = (const float*)d_in[10];
    const float* be    = (const float*)d_in[11];
    const float* Wm    = (const float*)d_in[12];
    const float* bm    = (const float*)d_in[13];
    const int* enc_len = (const int*)d_in[14];

    char* ws = (char*)d_ws;
    unsigned short* wkb   = (unsigned short*)(ws + 0);           // 1 MB
    unsigned short* wvb   = (unsigned short*)(ws + 1048576);     // 2 MB
    float*          qbuf  = (float*)(ws + 3145728);              // 128 KB
    unsigned short* locb  = (unsigned short*)(ws + 3276800);     // 32 MB
    float*          ebuf  = (float*)(ws + 36831232);             // 1 MB
    float*          cbuf  = (float*)(ws + 37879808);             // 256 KB

    float* attn_out = (float*)d_out;                 // [32][4][2048]
    float* context_out = attn_out + BS * NH * TS;    // [32][512]

    prep_kernel<<<512, 256, 0, stream>>>(Wk, Wv, wkb, wvb);
    q_kernel<<<128, 256, 0, stream>>>(dec, Wq, bq, qbuf);
    loc_kernel<<<4096, 256, 0, stream>>>(Wconv, Wloc, enc_len, locb);
    energy_kernel<<<4096, 256, 0, stream>>>(enc, wkb, bk, qbuf, locb, We, be, ebuf);
    softmax_kernel<<<128, 256, 0, stream>>>(ebuf, enc_len, attn_out);
    ctx_kernel<<<256, 256, 0, stream>>>(enc, wvb, bv, attn_out, cbuf);
    merge_kernel<<<64, 256, 0, stream>>>(cbuf, Wm, bm, context_out);
}
